// AOAReader_Modeler_31679678775975
// MI455X (gfx1250) — matrix operand order repaired and checked by reading, not compiled, not run on hardware
//
#include <hip/hip_runtime.h>
#include <hip/hip_bf16.h>
#include <cstdint>
#include <cstddef>

// ---------------------------------------------------------------------------
// Problem constants (from reference)
// ---------------------------------------------------------------------------
#define BB   32
#define LC   1024
#define LQ   48
#define HH   128
#define EE   128
#define KK   20
#define G3   384      // 3*H
#define N768 768      // both directions' gates concatenated
#define VV   50000

typedef __bf16 bf16;
typedef __attribute__((ext_vector_type(16))) __bf16 v16bf;
typedef __attribute__((ext_vector_type(8)))  float  v8f;

// ---------------------------------------------------------------------------
// WMMA helpers (CDNA5 16x16x32 bf16, f32 accumulate)
// ---------------------------------------------------------------------------
__device__ __forceinline__ v8f vzero8() {
  v8f z;
#pragma unroll
  for (int i = 0; i < 8; ++i) z[i] = 0.0f;
  return z;
}

__device__ __forceinline__ v8f wmma_bf16(v16bf a, v16bf b, v8f c) {
  // 8 args: (neg_a, A, neg_b, B, c_mod, C, reuse_a, reuse_b)
  return __builtin_amdgcn_wmma_f32_16x16x32_bf16(false, a, false, b,
                                                 (short)0, c, false, false);
}

// A fragment: 16(M) x 32(K) bf16 tile from row-major memory, row stride ld.
__device__ __forceinline__ v16bf load_a_frag(const bf16* src, int ld, int lane) {
  int r  = lane & 15;
  int kb = (lane >> 4) << 3;          // 0 or 8
  const bf16* p = src + r * ld + kb;
  v16bf a;
#pragma unroll
  for (int i = 0; i < 8; ++i) a[i] = p[i];
#pragma unroll
  for (int i = 0; i < 8; ++i) a[8 + i] = p[16 + i];
  return a;
}

// B fragment: 32(K) x 16(N) bf16 tile from row-major [K][N] memory, stride ld.
__device__ __forceinline__ v16bf load_b_frag(const bf16* src, int ld, int lane) {
  int n  = lane & 15;
  int kb = (lane >> 4) << 3;          // 0 or 8
  const bf16* p = src + kb * ld + n;
  v16bf b;
#pragma unroll
  for (int i = 0; i < 8; ++i) b[i] = p[i * ld];
#pragma unroll
  for (int i = 0; i < 8; ++i) b[8 + i] = p[(16 + i) * ld];
  return b;
}

// D fragment store: 16x16 f32 -> row-major dst with row stride ld.
__device__ __forceinline__ void store_d(float* dst, int ld, v8f d, int lane) {
  int n  = lane & 15;
  int m0 = (lane >> 4) << 3;
#pragma unroll
  for (int i = 0; i < 8; ++i) dst[(m0 + i) * ld + n] = d[i];
}

__device__ __forceinline__ float sigm_f(float x) {
  return 1.0f / (1.0f + __expf(-x));
}
__device__ __forceinline__ float tanh_f(float x) {
  float e = __expf(-2.0f * x);
  return (1.0f - e) / (1.0f + e);
}

// ---------------------------------------------------------------------------
// CDNA5 async global->LDS copy (ASYNCcnt-tracked, per-lane 16B)
// ---------------------------------------------------------------------------
__device__ __forceinline__ void async_b128(const void* gptr, void* lds_generic) {
  uint32_t lo = (uint32_t)(uintptr_t)lds_generic;          // low 32b = LDS offset
  unsigned long long g = (unsigned long long)(uintptr_t)gptr;
  asm volatile("global_load_async_to_lds_b128 %0, %1, off"
               :: "v"(lo), "v"(g) : "memory");
}
__device__ __forceinline__ void wait_async_le6() {
  asm volatile("s_wait_asynccnt 0x6" ::: "memory");
}
__device__ __forceinline__ void wait_async_0() {
  asm volatile("s_wait_asynccnt 0x0" ::: "memory");
}

// ---------------------------------------------------------------------------
// Kernel 0: convert embedding table to bf16 (halves gather traffic; enables
// direct async global->LDS staging).
// ---------------------------------------------------------------------------
__global__ void conv_emb(const float* __restrict__ emb, bf16* __restrict__ out) {
  size_t i = (size_t)blockIdx.x * 256 + threadIdx.x;
  if (i < (size_t)VV * EE) out[i] = (bf16)emb[i];
}

// ---------------------------------------------------------------------------
// Kernel 1: repack weights to bf16 WMMA-friendly transposed layouts.
//   WihT[k*768 + d*384 + n] = Wih[d][n][k]      (k in [0,128), concat dirs)
//   WhhT[d][k*384 + n]      = Whh[d][n][k]
// ---------------------------------------------------------------------------
__global__ void prep_weights(const float* __restrict__ ctxWih,
                             const float* __restrict__ ctxWhh,
                             const float* __restrict__ qstWih,
                             const float* __restrict__ qstWhh,
                             bf16* __restrict__ WihT_ctx, bf16* __restrict__ WihT_qst,
                             bf16* __restrict__ WhhT_ctx, bf16* __restrict__ WhhT_qst) {
  int idx = blockIdx.x * blockDim.x + threadIdx.x;
  if (idx >= 128 * N768) return;
  {
    int k = idx / N768, c = idx % N768;                 // c = d*384 + n
    WihT_ctx[idx] = (bf16)ctxWih[c * 128 + k];
    WihT_qst[idx] = (bf16)qstWih[c * 128 + k];
  }
  {
    int d = idx / (128 * G3), r = idx % (128 * G3);
    int k = r / G3, n = r % G3;
    WhhT_ctx[idx] = (bf16)ctxWhh[(d * G3 + n) * 128 + k];
    WhhT_qst[idx] = (bf16)qstWhh[(d * G3 + n) * 128 + k];
  }
}

// ---------------------------------------------------------------------------
// Kernel 2: fused embedding gather + input projection GEMM.
// Row gather staged via async global->LDS (bf16 table), then WMMA.
// Block: 256 threads (8 waves), one 16-row M tile, N=768 (6 n-tiles/wave:
// 48 acc VGPRs/wave -> no spills).
// ---------------------------------------------------------------------------
__global__ void embed_proj(const int* __restrict__ tokens,
                           const bf16* __restrict__ embb,
                           const bf16* __restrict__ WihT,
                           const float* __restrict__ bih,
                           float* __restrict__ xW) {
  __shared__ __attribute__((aligned(16))) bf16 As[16 * 128];
  int m0   = blockIdx.x * 16;
  int tid  = threadIdx.x;
  int lane = tid & 31, wave = tid >> 5;

  // 16 rows x 256B = 256 chunks of 16B; exactly 1 async b128 per thread
  {
    int c = tid;
    int m = c >> 4;
    int o16 = (c & 15) * 8;                 // bf16 element offset in row
    int tok = tokens[m0 + m];
    async_b128(embb + (size_t)tok * EE + o16, &As[m * 128 + o16]);
  }
  wait_async_0();
  __syncthreads();

  v8f acc[6];
#pragma unroll
  for (int j = 0; j < 6; ++j) acc[j] = vzero8();

#pragma unroll
  for (int kt = 0; kt < 4; ++kt) {
    v16bf a = load_a_frag(As + kt * 32, 128, lane);
#pragma unroll
    for (int j = 0; j < 6; ++j) {
      int nt = wave * 6 + j;
      v16bf b = load_b_frag(WihT + (kt * 32) * N768 + nt * 16, N768, lane);
      acc[j] = wmma_bf16(a, b, acc[j]);
    }
  }

  int n  = lane & 15;
  int mr = (lane >> 4) << 3;
#pragma unroll
  for (int j = 0; j < 6; ++j) {
    int nt = wave * 6 + j;
    float bia = bih[nt * 16 + n];
    float* dst = xW + (size_t)m0 * N768 + nt * 16;
#pragma unroll
    for (int i = 0; i < 8; ++i)
      dst[(size_t)(mr + i) * N768 + n] = acc[j][i] + bia;
  }
}

// ---------------------------------------------------------------------------
// Kernel 3: persistent GRU scan. grid = 4 blocks: (dir, batch-half).
// 256 threads (8 waves). Whh lives in registers as 12 B-fragments/wave,
// h ping-pongs in LDS, and xW for step t+1 is async-prefetched into an LDS
// double buffer while step t computes (ASYNCcnt pipelining).
// ---------------------------------------------------------------------------
__global__ void gru_scan(const float* __restrict__ xW,
                         const bf16* __restrict__ WhhT2,   // [2][128][384]
                         const float* __restrict__ bhh2,   // [2][384]
                         const float* __restrict__ h02,    // [2][128]
                         bf16* __restrict__ outp,
                         int L, int sB, int sT, int sJ) {
  int dir   = blockIdx.x >> 1;
  int mbase = (blockIdx.x & 1) << 4;
  const bf16*  WhhT = WhhT2 + (size_t)dir * 128 * G3;
  const float* bhh  = bhh2 + dir * G3;
  const float* h0   = h02 + dir * HH;

  __shared__ __attribute__((aligned(16))) bf16  hS[2][16 * 128];
  __shared__ __attribute__((aligned(16))) float ghS[16 * G3];
  __shared__ __attribute__((aligned(16))) float xwS[2][16 * G3];

  int tid = threadIdx.x;
  int lane = tid & 31, wave = tid >> 5;

  // Persistent Whh B-fragments: 12 x v16bf = 96 VGPRs.
  v16bf bf[3][4];
#pragma unroll
  for (int j = 0; j < 3; ++j)
#pragma unroll
    for (int kt = 0; kt < 4; ++kt) {
      int nt = wave * 3 + j;
      bf[j][kt] = load_b_frag(WhhT + (kt * 32) * G3 + nt * 16, G3, lane);
    }

  for (int i = tid; i < 16 * 128; i += 256) hS[0][i] = (bf16)h0[i & 127];

  // Prologue: async-prefetch xW tile for t=0 into buffer 0.
  // Tile = 16 rows x 384 f32 = 1536 chunks of 16B; 6 per thread.
  {
    int xt0 = dir ? (L - 1) : 0;
#pragma unroll
    for (int r = 0; r < 6; ++r) {
      int c = tid + r * 256;
      int m = c / 96;
      int co = (c - m * 96) * 4;            // f32 offset within row
      const float* g = xW + ((size_t)(mbase + m) * L + xt0) * N768 + dir * G3 + co;
      async_b128(g, &xwS[0][c * 4]);
    }
  }
  __syncthreads();

  int p = 0;
  for (int t = 0; t < L; ++t) {
    // gh = h @ Whh^T  (per wave: 3 n-tiles x 4 k-tiles = 12 WMMAs)
    v16bf a[4];
#pragma unroll
    for (int kt = 0; kt < 4; ++kt)
      a[kt] = load_a_frag(&hS[p][kt * 32], 128, lane);
#pragma unroll
    for (int j = 0; j < 3; ++j) {
      v8f acc = vzero8();
#pragma unroll
      for (int kt = 0; kt < 4; ++kt) acc = wmma_bf16(a[kt], bf[j][kt], acc);
      store_d(ghS + (wave * 3 + j) * 16, G3, acc, lane);
    }

    // Kick off next step's xW prefetch, then retire current buffer
    // (ASYNCcnt completes in order: wait<=6 retires the 6 oldest).
    if (t + 1 < L) {
      int xtn = dir ? (L - 2 - t) : (t + 1);
#pragma unroll
      for (int r = 0; r < 6; ++r) {
        int c = tid + r * 256;
        int m = c / 96;
        int co = (c - m * 96) * 4;
        const float* g = xW + ((size_t)(mbase + m) * L + xtn) * N768 + dir * G3 + co;
        async_b128(g, &xwS[p ^ 1][c * 4]);
      }
      wait_async_le6();
    } else {
      wait_async_0();
    }
    __syncthreads();

    int xt = dir ? (L - 1 - t) : t;
    const float* xwp = xwS[p];
#pragma unroll
    for (int i = 0; i < 8; ++i) {
      int e = tid + i * 256;            // 0..2047, coalesced in j
      int m = e >> 7, j = e & 127;
      int b = mbase + m;
      float ghr = ghS[m * G3 + j]       + bhh[j];
      float ghz = ghS[m * G3 + 128 + j] + bhh[128 + j];
      float ghn = ghS[m * G3 + 256 + j] + bhh[256 + j];
      float rr = sigm_f(xwp[m * G3 + j]       + ghr);
      float zz = sigm_f(xwp[m * G3 + 128 + j] + ghz);
      float nn = tanh_f(xwp[m * G3 + 256 + j] + rr * ghn);
      float hold = (float)hS[p][m * 128 + j];
      float h = (1.0f - zz) * nn + zz * hold;
      hS[p ^ 1][m * 128 + j] = (bf16)h;
      outp[(size_t)b * sB + (size_t)xt * sT + (dir * 128 + j) * sJ] = (bf16)h;
    }
    __syncthreads();
    p ^= 1;
  }
}

// ---------------------------------------------------------------------------
// Kernel 4: attention scores  M[b][l][q] = sum_j C[b][l][j] * Qt[b][j][q]
// Block: 128 threads (4 waves), 64 l-rows per block; grid (LC/64, B).
// ---------------------------------------------------------------------------
__global__ void attn_scores(const bf16* __restrict__ C,
                            const bf16* __restrict__ Qt,
                            float* __restrict__ M) {
  int b  = blockIdx.y;
  int l0 = blockIdx.x * 64;
  int tid = threadIdx.x;
  int lane = tid & 31, wave = tid >> 5;

  const bf16* A  = C + ((size_t)b * LC + l0 + wave * 16) * 256;
  const bf16* Bm = Qt + (size_t)b * 256 * LQ;

  v8f acc[3];
#pragma unroll
  for (int j = 0; j < 3; ++j) acc[j] = vzero8();

#pragma unroll
  for (int kt = 0; kt < 8; ++kt) {
    v16bf a = load_a_frag(A + kt * 32, 256, lane);
#pragma unroll
    for (int j = 0; j < 3; ++j) {
      v16bf bb = load_b_frag(Bm + (kt * 32) * LQ + j * 16, LQ, lane);
      acc[j] = wmma_bf16(a, bb, acc[j]);
    }
  }
  float* D = M + ((size_t)b * LC + l0 + wave * 16) * LQ;
#pragma unroll
  for (int j = 0; j < 3; ++j) store_d(D + j * 16, LQ, acc[j], lane);
}

// ---------------------------------------------------------------------------
// Kernel 5: per-row softmax over q, av partials (deterministic LDS tree).
// ---------------------------------------------------------------------------
__global__ void row_soft(const float* __restrict__ M, float* __restrict__ avpart) {
  __shared__ float buf[256][LQ];
  int b = blockIdx.y, chunk = blockIdx.x;
  int tid = threadIdx.x;
  int l = chunk * 256 + tid;
  const float* row = M + ((size_t)b * LC + l) * LQ;

  float v[LQ];
  float mx = -1e30f;
#pragma unroll
  for (int q = 0; q < LQ; ++q) { v[q] = row[q]; mx = fmaxf(mx, v[q]); }
  float s = 0.0f;
#pragma unroll
  for (int q = 0; q < LQ; ++q) { v[q] = __expf(v[q] - mx); s += v[q]; }
  float inv = 1.0f / (s * (float)LC);
#pragma unroll
  for (int q = 0; q < LQ; ++q) buf[tid][q] = v[q] * inv;
  __syncthreads();
  if (tid < LQ) {
    float s2 = 0.0f;
    for (int i = 0; i < 256; ++i) s2 += buf[i][tid];
    avpart[(b * 4 + chunk) * LQ + tid] = s2;
  }
}

__global__ void av_reduce(const float* __restrict__ avpart, float* __restrict__ av) {
  int idx = blockIdx.x * blockDim.x + threadIdx.x;
  if (idx >= BB * LQ) return;
  int b = idx / LQ, q = idx % LQ;
  float s = 0.0f;
#pragma unroll
  for (int c = 0; c < 4; ++c) s += avpart[(b * 4 + c) * LQ + q];
  av[idx] = s;
}

// ---------------------------------------------------------------------------
// Kernel 6: column (context-dim) softmax stats per (b,q). grid (LQ, B).
// ---------------------------------------------------------------------------
__global__ void col_stats(const float* __restrict__ M,
                          float* __restrict__ colmax, float* __restrict__ colsum) {
  __shared__ float red[128];
  int b = blockIdx.y, q = blockIdx.x;
  int tid = threadIdx.x;
  const float* base = M + (size_t)b * LC * LQ + q;
  float mx = -1e30f;
  for (int l = tid; l < LC; l += 128) mx = fmaxf(mx, base[(size_t)l * LQ]);
  red[tid] = mx; __syncthreads();
  for (int s = 64; s > 0; s >>= 1) {
    if (tid < s) red[tid] = fmaxf(red[tid], red[tid + s]);
    __syncthreads();
  }
  mx = red[0]; __syncthreads();
  float sm = 0.0f;
  for (int l = tid; l < LC; l += 128) sm += __expf(base[(size_t)l * LQ] - mx);
  red[tid] = sm; __syncthreads();
  for (int s = 64; s > 0; s >>= 1) {
    if (tid < s) red[tid] += red[tid + s];
    __syncthreads();
  }
  if (tid == 0) { colmax[b * LQ + q] = mx; colsum[b * LQ + q] = red[0]; }
}

// ---------------------------------------------------------------------------
// Kernel 7: o[b][l] = sum_q cols_att[b][l][q] * av[b][q]
// ---------------------------------------------------------------------------
__global__ void calc_o(const float* __restrict__ M, const float* __restrict__ colmax,
                       const float* __restrict__ colsum, const float* __restrict__ av,
                       float* __restrict__ o) {
  int idx = blockIdx.x * blockDim.x + threadIdx.x;   // b*LC + l
  if (idx >= BB * LC) return;
  int b = idx >> 10;
  const float* row = M + (size_t)idx * LQ;
  const float* cm = colmax + b * LQ;
  const float* cs = colsum + b * LQ;
  const float* ab = av + b * LQ;
  float s = 0.0f;
#pragma unroll
  for (int q = 0; q < LQ; ++q)
    s += __expf(row[q] - cm[q]) / cs[q] * ab[q];
  o[idx] = s;
}

// ---------------------------------------------------------------------------
// Kernel 8: candidate aggregation + per-b log-softmax/loss/acc. grid (B).
// ---------------------------------------------------------------------------
__global__ void cand_final(const int* __restrict__ context,
                           const int* __restrict__ candidates,
                           const int* __restrict__ target,
                           const float* __restrict__ o,
                           float* __restrict__ out,
                           float* __restrict__ lossb, float* __restrict__ accb) {
  __shared__ float buf[256][KK];
  __shared__ float Ssh[KK];
  __shared__ float PW[KK];
  __shared__ int   candS[KK];
  __shared__ float lseS;
  int b = blockIdx.x, tid = threadIdx.x;

  if (tid < KK) candS[tid] = candidates[b * KK + tid];
  __syncthreads();

  int cnd[KK];
#pragma unroll
  for (int k = 0; k < KK; ++k) cnd[k] = candS[k];

  float S[KK];
#pragma unroll
  for (int k = 0; k < KK; ++k) S[k] = 0.0f;
  for (int l = tid; l < LC; l += 256) {
    int tok = context[b * LC + l];
    float ov = o[b * LC + l];
#pragma unroll
    for (int k = 0; k < KK; ++k)
      if (tok == cnd[k]) S[k] += ov;
  }
#pragma unroll
  for (int k = 0; k < KK; ++k) buf[tid][k] = S[k];
  __syncthreads();

  if (tid < KK) {
    float s = 0.0f;
    for (int i = 0; i < 256; ++i) s += buf[i][tid];
    Ssh[tid] = s;
  }
  __syncthreads();

  if (tid < KK) {
    int ck = candS[tid];
    float s = 0.0f;
#pragma unroll
    for (int j = 0; j < KK; ++j) {
      int cj = candS[j];
      if (ck == cj && cj != 0) s += Ssh[j];
    }
    PW[tid] = s;
  }
  __syncthreads();

  if (tid == 0) {
    float mx = -1e30f; int am = 0;
#pragma unroll
    for (int k = 0; k < KK; ++k)
      if (PW[k] > mx) { mx = PW[k]; am = k; }
    float se = 0.0f;
#pragma unroll
    for (int k = 0; k < KK; ++k) se += __expf(PW[k] - mx);
    float lse = mx + __logf(se);
    lseS = lse;
    int tg = target[b];
    lossb[b] = -(PW[tg] - lse);
    accb[b] = (am == tg) ? 1.0f : 0.0f;
  }
  __syncthreads();

  if (tid < KK) {
    float ls = PW[tid] - lseS;
    out[2 + b * KK + tid] = ls;                    // log_soft
    out[2 + BB * KK + b * KK + tid] = __expf(ls);  // soft
  }
}

__global__ void finalize(const float* __restrict__ lossb,
                         const float* __restrict__ accb, float* __restrict__ out) {
  if (threadIdx.x == 0) {
    float l = 0.0f, a = 0.0f;
    for (int b = 0; b < BB; ++b) { l += lossb[b]; a += accb[b]; }
    out[0] = l / (float)BB;
    out[1] = a / (float)BB;
  }
}

// ---------------------------------------------------------------------------
// Host launcher
// ---------------------------------------------------------------------------
extern "C" void kernel_launch(void* const* d_in, const int* in_sizes, int n_in,
                              void* d_out, int out_size, void* d_ws, size_t ws_size,
                              hipStream_t stream) {
  const int*   context    = (const int*)d_in[0];
  const int*   question   = (const int*)d_in[1];
  const int*   candidates = (const int*)d_in[2];
  const int*   target     = (const int*)d_in[3];
  const float* emb        = (const float*)d_in[4];
  const float* ctx_Wih    = (const float*)d_in[5];
  const float* ctx_Whh    = (const float*)d_in[6];
  const float* ctx_bih    = (const float*)d_in[7];
  const float* ctx_bhh    = (const float*)d_in[8];
  const float* ctx_h0     = (const float*)d_in[9];
  const float* qst_Wih    = (const float*)d_in[10];
  const float* qst_Whh    = (const float*)d_in[11];
  const float* qst_bih    = (const float*)d_in[12];
  const float* qst_bhh    = (const float*)d_in[13];
  const float* qst_h0     = (const float*)d_in[14];
  float* out = (float*)d_out;

  char* base = (char*)d_ws;
  size_t off = 0;
  auto alloc = [&](size_t bytes) -> char* {
    off = (off + 255) & ~(size_t)255;
    char* p = base + off;
    off += bytes;
    return p;
  };

  bf16*  emb_bf   = (bf16*)alloc((size_t)VV * EE * 2);
  bf16*  WihT_ctx = (bf16*)alloc((size_t)128 * N768 * 2);
  bf16*  WihT_qst = (bf16*)alloc((size_t)128 * N768 * 2);
  bf16*  WhhT_ctx = (bf16*)alloc((size_t)2 * 128 * G3 * 2);
  bf16*  WhhT_qst = (bf16*)alloc((size_t)2 * 128 * G3 * 2);
  float* xW_ctx   = (float*)alloc((size_t)BB * LC * N768 * 4);
  float* xW_qst   = (float*)alloc((size_t)BB * LQ * N768 * 4);
  bf16*  Cbuf     = (bf16*)alloc((size_t)BB * LC * 256 * 2);
  bf16*  Qt       = (bf16*)alloc((size_t)BB * 256 * LQ * 2);
  float* Mbuf     = (float*)alloc((size_t)BB * LC * LQ * 4);
  float* avpart   = (float*)alloc((size_t)BB * 4 * LQ * 4);
  float* av       = (float*)alloc((size_t)BB * LQ * 4);
  float* colmax   = (float*)alloc((size_t)BB * LQ * 4);
  float* colsum   = (float*)alloc((size_t)BB * LQ * 4);
  float* obuf     = (float*)alloc((size_t)BB * LC * 4);
  float* lossb    = (float*)alloc((size_t)BB * 4);
  float* accb     = (float*)alloc((size_t)BB * 4);
  (void)ws_size; (void)in_sizes; (void)n_in; (void)out_size;

  // 0) embedding table -> bf16
  conv_emb<<<(VV * EE + 255) / 256, 256, 0, stream>>>(emb, emb_bf);

  // 1) weight repack to bf16 transposed layouts
  prep_weights<<<(128 * N768 + 255) / 256, 256, 0, stream>>>(
      ctx_Wih, ctx_Whh, qst_Wih, qst_Whh, WihT_ctx, WihT_qst, WhhT_ctx, WhhT_qst);

  // 2) fused embed + input projection (async LDS staging + WMMA GEMM)
  embed_proj<<<(BB * LC) / 16, 256, 0, stream>>>(context, emb_bf, WihT_ctx, ctx_bih, xW_ctx);
  embed_proj<<<(BB * LQ) / 16, 256, 0, stream>>>(question, emb_bf, WihT_qst, qst_bih, xW_qst);

  // 3) BiGRU scans (persistent WMMA recurrence + async xW double buffer)
  gru_scan<<<4, 256, 0, stream>>>(xW_ctx, WhhT_ctx, ctx_bhh, ctx_h0, Cbuf,
                                  LC, LC * 256, 256, 1);
  gru_scan<<<4, 256, 0, stream>>>(xW_qst, WhhT_qst, qst_bhh, qst_h0, Qt,
                                  LQ, 256 * LQ, 1, LQ);

  // 4) attention scores M = C . Q^T (WMMA GEMM)
  attn_scores<<<dim3(LC / 64, BB), 128, 0, stream>>>(Cbuf, Qt, Mbuf);

  // 5) AoA: row softmax -> av ; column stats -> o
  row_soft<<<dim3(4, BB), 256, 0, stream>>>(Mbuf, avpart);
  av_reduce<<<(BB * LQ + 255) / 256, 256, 0, stream>>>(avpart, av);
  col_stats<<<dim3(LQ, BB), 128, 0, stream>>>(Mbuf, colmax, colsum);
  calc_o<<<(BB * LC) / 256, 256, 0, stream>>>(Mbuf, colmax, colsum, av, obuf);

  // 6) candidate aggregation + outputs
  cand_final<<<BB, 256, 0, stream>>>(context, candidates, target, obuf, out, lossb, accb);
  finalize<<<1, 32, 0, stream>>>(lossb, accb, out);
}
